// Hopfield_18992345383067
// MI455X (gfx1250) — matrix operand order repaired and checked by reading, not compiled, not run on hardware
//
#include <hip/hip_runtime.h>
#include <math.h>

// ---------------- problem constants ----------------
constexpr int kE  = 1024;   // embedding
constexpr int kH  = 16;     // heads
constexpr int kHD = 64;     // head dim
constexpr int kS  = 1024;   // sequence
constexpr int kB  = 8;      // batch
constexpr float kLnEps = 1e-5f;
constexpr float kScale = 0.125f;      // HD^-0.5

typedef __attribute__((ext_vector_type(16))) _Float16 v16h;
typedef __attribute__((ext_vector_type(8)))  float    v8f;

// ---- CDNA5 async global->LDS copy (signature per hipcc diagnostic:
//      (v4i addrspace(1)*, v4i addrspace(3)*, imm offset, imm cpol)) ----
typedef int v4i_vs __attribute__((vector_size(16)));

#if defined(__HIP_DEVICE_COMPILE__) && __has_builtin(__builtin_amdgcn_global_load_async_to_lds_b128)
#define HAVE_ASYNC_LDS 1
#define ASYNC_CP_B128(gsrc, ldst)                                              \
  __builtin_amdgcn_global_load_async_to_lds_b128(                              \
      (__attribute__((address_space(1))) v4i_vs*)(gsrc),                       \
      (__attribute__((address_space(3))) v4i_vs*)(ldst), 0, 0)
#endif

__device__ __forceinline__ void wait_async0() {
#if __has_builtin(__builtin_amdgcn_s_wait_asynccnt)
  __builtin_amdgcn_s_wait_asynccnt(0);
#else
  asm volatile("s_wait_asynccnt 0x0" ::: "memory");
#endif
}

__device__ __forceinline__ v8f wmma16x16x32(v16h a, v16h b, v8f c) {
  return __builtin_amdgcn_wmma_f32_16x16x32_f16(
      /*neg_a=*/false, a, /*neg_b=*/false, b,
      /*c_mod=*/(short)0, c, /*reuse_a=*/false, /*reuse_b=*/false);
}

// reductions within a 16-lane half of the wave (C/D rows live in half-waves)
__device__ __forceinline__ float half16_sum(float v) {
  v += __shfl_xor(v, 1, 32);
  v += __shfl_xor(v, 2, 32);
  v += __shfl_xor(v, 4, 32);
  v += __shfl_xor(v, 8, 32);
  return v;
}
__device__ __forceinline__ float half16_max(float v) {
  v = fmaxf(v, __shfl_xor(v, 1, 32));
  v = fmaxf(v, __shfl_xor(v, 2, 32));
  v = fmaxf(v, __shfl_xor(v, 4, 32));
  v = fmaxf(v, __shfl_xor(v, 8, 32));
  return v;
}

// =====================================================================
// Kernel 1: fused QKV projection + bias + LayerNorm, f32 in -> f16 heads out.
// grid = (48 n-groups of 64 weight rows, 64 m-blocks of 128 rows), 128 thr.
// Each wave owns a 32(M)x64(N=one head) tile (8 WMMAs per K-step against
// 4 shared B fragments); the 4 waves share the weight tile staged in LDS.
// =====================================================================
__global__ __launch_bounds__(128)
void proj_ln_kernel(const float* __restrict__ qx,
                    const float* __restrict__ kx,
                    const float* __restrict__ vx,
                    const float* __restrict__ W,     // (3E, E) row-major
                    const float* __restrict__ bias,  // (3E)
                    _Float16* __restrict__ qo,       // (B,H,S,HD)
                    _Float16* __restrict__ ko,
                    _Float16* __restrict__ vo) {
  __shared__ _Float16 Bs[64][32];     // weight tile, n-major [n][k]

  const int tid  = threadIdx.x;
  const int lane = tid & 31;
  const int wid  = tid >> 5;
  const int g    = blockIdx.x;        // 0..47
  const int mat  = g >> 4;            // 0=q,1=k,2=v
  const int h    = g & 15;            // head within matrix
  const float*   x = (mat == 0) ? qx : ((mat == 1) ? kx : vx);
  _Float16*      o = (mat == 0) ? qo : ((mat == 1) ? ko : vo);
  const int nbase = g * 64;           // global weight row
  const int m0    = blockIdx.y * 128 + wid * 32;
  const int row   = lane & 15;
  const int kh    = lane >> 4;
  const int col   = lane & 15;

  v8f acc[2][4] = {};

  const int sn  = tid >> 1;           // staging: weight row
  const int ssh = (tid & 1) * 16;     // staging: k segment

  for (int kb = 0; kb < kE; kb += 32) {
    __syncthreads();
    {   // stage 64x32 weight tile: 2 threads per n-row, 16 contiguous f32 each
      const float* src = W + (size_t)(nbase + sn) * kE + kb + ssh;
      #pragma unroll
      for (int j = 0; j < 16; ++j) Bs[sn][ssh + j] = (_Float16)src[j];
      if (kb + 32 < kE) __builtin_prefetch(src + 32, 0, 3);
    }
    __syncthreads();

    // A fragments (two 16-row tiles) from f32 activations
    v16h a[2];
    #pragma unroll
    for (int mt = 0; mt < 2; ++mt) {
      const float* ap = x + (size_t)(m0 + mt * 16 + row) * kE + kb + kh * 16;
      #pragma unroll
      for (int e = 0; e < 16; ++e) a[mt][e] = (_Float16)ap[e];
    }
    // preload all B fragments, then burst the 8 WMMAs
    v16h b[4];
    #pragma unroll
    for (int nt = 0; nt < 4; ++nt) {
      const _Float16* bp = &Bs[nt * 16 + col][kh * 16];
      #pragma unroll
      for (int e = 0; e < 16; ++e) b[nt][e] = bp[e];
    }
    #pragma unroll
    for (int mt = 0; mt < 2; ++mt)
      #pragma unroll
      for (int nt = 0; nt < 4; ++nt)
        acc[mt][nt] = wmma16x16x32(a[mt], b[nt], acc[mt][nt]);
  }

  // bias + LayerNorm over the 64-wide head row, scatter to (B,H,S,HD) as f16
  float bl[4];
  #pragma unroll
  for (int nt = 0; nt < 4; ++nt) bl[nt] = bias[nbase + nt * 16 + col];

  #pragma unroll
  for (int mt = 0; mt < 2; ++mt) {
    #pragma unroll
    for (int r = 0; r < 8; ++r) {
      float xs[4];
      float s1 = 0.f, s2 = 0.f;
      #pragma unroll
      for (int nt = 0; nt < 4; ++nt) {
        xs[nt] = acc[mt][nt][r] + bl[nt];
        s1 += xs[nt];
        s2 += xs[nt] * xs[nt];
      }
      s1 = half16_sum(s1);
      s2 = half16_sum(s2);
      const float mean = s1 * (1.0f / 64.0f);
      const float var  = s2 * (1.0f / 64.0f) - mean * mean;
      const float rstd = rsqrtf(var + kLnEps);
      const int m  = m0 + mt * 16 + r + 8 * kh;
      const int s  = m / kB;
      const int bi = m % kB;
      const size_t ob = (((size_t)(bi * kH + h)) * kS + s) * kHD;
      #pragma unroll
      for (int nt = 0; nt < 4; ++nt)
        o[ob + nt * 16 + col] = (_Float16)((xs[nt] - mean) * rstd);
    }
  }
}

// =====================================================================
// Kernel 2: one Hopfield attention pass with flash-style online softmax.
//   scores = softmax(Q Kᵀ * 0.125) ; out = scores @ Vm
//   final_pass==0: Vm==K, output layernormed f16 q_next in (B,H,S,HD)
//   final_pass==1: Vm==V, output f16 attention result in (S,B,E)
// grid = (B*H = 128, S/64 = 16), 128 threads; each wave owns 16 q-rows.
// K chunk staged row-major via async-to-LDS; V chunk staged TRANSPOSED
// so GEMM2 B-fragments are contiguous ds_load_b128.
// =====================================================================
__global__ __launch_bounds__(128)
void attn_pass_kernel(const _Float16* __restrict__ Q,
                      const _Float16* __restrict__ K,
                      const _Float16* __restrict__ Vm,
                      _Float16* __restrict__ Out,
                      int final_pass) {
  __shared__ _Float16 Ks[32][kHD];      // K chunk (32 keys x 64), row-major
  __shared__ _Float16 VsT[kHD][32];     // V chunk transposed [d][key]
  __shared__ _Float16 Ps[4][16][32];    // per-wave probability tile (A-relayout)

  const int tid  = threadIdx.x;
  const int lane = tid & 31;
  const int wid  = tid >> 5;
  const int bh   = blockIdx.x;
  const int q0r  = blockIdx.y * 64 + wid * 16;
  const _Float16* Qh = Q  + (size_t)bh * kS * kHD;
  const _Float16* Kh = K  + (size_t)bh * kS * kHD;
  const _Float16* Vh = Vm + (size_t)bh * kS * kHD;
  const int row = lane & 15;
  const int kh  = lane >> 4;
  const int col = lane & 15;

  // Q A-fragments for both HD halves are loop-invariant: load once.
  v16h aq0, aq1;
  {
    const _Float16* ap = Qh + (size_t)(q0r + row) * kHD + kh * 16;
    #pragma unroll
    for (int e = 0; e < 16; ++e) {
      const int off = e;
      aq0[e] = ap[off];
      aq1[e] = ap[32 + off];
    }
  }

  v8f o[4] = {};
  float rm[8], rl[8];
  #pragma unroll
  for (int r = 0; r < 8; ++r) { rm[r] = -1e30f; rl[r] = 0.f; }

  const int rr  = tid >> 2;           // staging: key row (0..31)
  const int seg = (tid & 3) * 16;     // staging: d segment

  for (int kc = 0; kc < kS; kc += 32) {
    __syncthreads();
    const _Float16* ksrc = Kh + (size_t)(kc + rr) * kHD + seg;
    const _Float16* vsrc = Vh + (size_t)(kc + rr) * kHD + seg;
#ifdef HAVE_ASYNC_LDS
    ASYNC_CP_B128(ksrc,     &Ks[rr][seg]);
    ASYNC_CP_B128(ksrc + 8, &Ks[rr][seg + 8]);
#else
    #pragma unroll
    for (int j = 0; j < 16; ++j) Ks[rr][seg + j] = ksrc[j];
#endif
    // V staged transposed: coalesced global read, scattered LDS write
    #pragma unroll
    for (int j = 0; j < 16; ++j) VsT[seg + j][rr] = vsrc[j];
    if (kc + 32 < kS) {
      __builtin_prefetch(ksrc + 32 * kHD, 0, 3);
      __builtin_prefetch(vsrc + 32 * kHD, 0, 3);
    }
#ifdef HAVE_ASYNC_LDS
    wait_async0();
#endif
    __syncthreads();

    // ---- GEMM1: scores 16x32 = Q(16x64) @ Kchunkᵀ(64x32) ----
    v16h kb0[2], kb1[2];
    #pragma unroll
    for (int nt = 0; nt < 2; ++nt) {
      const _Float16* bp = &Ks[nt * 16 + col][kh * 16];   // B n-major: Ks[key][d]
      #pragma unroll
      for (int e = 0; e < 16; ++e) { kb0[nt][e] = bp[e]; kb1[nt][e] = bp[32 + e]; }
    }
    v8f sc[2];
    #pragma unroll
    for (int nt = 0; nt < 2; ++nt) {
      v8f z = {};
      z = wmma16x16x32(aq0, kb0[nt], z);
      sc[nt] = wmma16x16x32(aq1, kb1[nt], z);
    }

    // ---- online softmax update ----
    float f[8];
    #pragma unroll
    for (int r = 0; r < 8; ++r) {
      const float s0 = sc[0][r] * kScale;
      const float s1 = sc[1][r] * kScale;
      const float cm = half16_max(fmaxf(s0, s1));
      const float nm = fmaxf(rm[r], cm);
      f[r]  = __expf(rm[r] - nm);
      rm[r] = nm;
      const float p0 = __expf(s0 - nm);
      const float p1 = __expf(s1 - nm);
      rl[r] = rl[r] * f[r] + half16_sum(p0 + p1);
      Ps[wid][r + 8 * kh][col]      = (_Float16)p0;   // relayout D -> A via LDS
      Ps[wid][r + 8 * kh][16 + col] = (_Float16)p1;
    }
    #pragma unroll
    for (int nt = 0; nt < 4; ++nt)
      #pragma unroll
      for (int r = 0; r < 8; ++r) o[nt][r] *= f[r];

    __syncthreads();   // make Ps (and keep VsT) ordered across the block

    // ---- GEMM2: o += P(16x32) @ Vchunk(32x64) ----
    v16h vb[4];
    #pragma unroll
    for (int nt = 0; nt < 4; ++nt) {
      const _Float16* bp = &VsT[nt * 16 + col][kh * 16];  // contiguous 16 f16
      #pragma unroll
      for (int e = 0; e < 16; ++e) vb[nt][e] = bp[e];
    }
    v16h ap;
    const _Float16* app = &Ps[wid][row][kh * 16];
    #pragma unroll
    for (int e = 0; e < 16; ++e) ap[e] = app[e];
    #pragma unroll
    for (int nt = 0; nt < 4; ++nt)
      o[nt] = wmma16x16x32(ap, vb[nt], o[nt]);
  }

  float rinv[8];
  #pragma unroll
  for (int r = 0; r < 8; ++r) rinv[r] = 1.0f / rl[r];

  if (final_pass) {
    // write attention output as (S, B, E) with e = h*64 + d
    const int bi = bh >> 4;
    const int h  = bh & 15;
    #pragma unroll
    for (int r = 0; r < 8; ++r) {
      const int s = q0r + r + 8 * kh;
      const size_t ob = ((size_t)s * kB + bi) * kE + (size_t)h * kHD;
      #pragma unroll
      for (int nt = 0; nt < 4; ++nt)
        Out[ob + nt * 16 + col] = (_Float16)(o[nt][r] * rinv[r]);
    }
  } else {
    // q_next = layernorm(attn @ K), stored (B,H,S,HD)
    #pragma unroll
    for (int r = 0; r < 8; ++r) {
      float xs[4];
      float s1 = 0.f, s2 = 0.f;
      #pragma unroll
      for (int nt = 0; nt < 4; ++nt) {
        xs[nt] = o[nt][r] * rinv[r];
        s1 += xs[nt];
        s2 += xs[nt] * xs[nt];
      }
      s1 = half16_sum(s1);
      s2 = half16_sum(s2);
      const float mean = s1 * (1.0f / 64.0f);
      const float var  = s2 * (1.0f / 64.0f) - mean * mean;
      const float rstd = rsqrtf(var + kLnEps);
      const int s = q0r + r + 8 * kh;
      const size_t ob = ((size_t)bh * kS + s) * kHD;
      #pragma unroll
      for (int nt = 0; nt < 4; ++nt)
        Out[ob + nt * 16 + col] = (_Float16)((xs[nt] - mean) * rstd);
    }
  }
}

// =====================================================================
// Kernel 3: output projection  out(8192x1024) = AO(f16) @ Woutᵀ + bias (f32)
// grid = (16 n-groups, 64 m-blocks of 128 rows), 128 threads,
// 32 M-rows per wave like the input projection.
// =====================================================================
__global__ __launch_bounds__(128)
void outproj_kernel(const _Float16* __restrict__ A,  // (M, E) f16
                    const float* __restrict__ W,     // (E, E) row-major
                    const float* __restrict__ bias,  // (E)
                    float* __restrict__ out) {       // (M, E) f32
  __shared__ _Float16 Bs[64][32];

  const int tid  = threadIdx.x;
  const int lane = tid & 31;
  const int wid  = tid >> 5;
  const int nbase = blockIdx.x * 64;
  const int m0    = blockIdx.y * 128 + wid * 32;
  const int row = lane & 15;
  const int kh  = lane >> 4;
  const int col = lane & 15;

  v8f acc[2][4] = {};

  const int sn  = tid >> 1;
  const int ssh = (tid & 1) * 16;

  for (int kb = 0; kb < kE; kb += 32) {
    __syncthreads();
    {
      const float* src = W + (size_t)(nbase + sn) * kE + kb + ssh;
      #pragma unroll
      for (int j = 0; j < 16; ++j) Bs[sn][ssh + j] = (_Float16)src[j];
      if (kb + 32 < kE) __builtin_prefetch(src + 32, 0, 3);
    }
    __syncthreads();

    v16h a[2];
    #pragma unroll
    for (int mt = 0; mt < 2; ++mt) {
      const _Float16* ap = A + (size_t)(m0 + mt * 16 + row) * kE + kb + kh * 16;
      #pragma unroll
      for (int e = 0; e < 16; ++e) a[mt][e] = ap[e];
    }
    v16h b[4];
    #pragma unroll
    for (int nt = 0; nt < 4; ++nt) {
      const _Float16* bp = &Bs[nt * 16 + col][kh * 16];
      #pragma unroll
      for (int e = 0; e < 16; ++e) b[nt][e] = bp[e];
    }
    #pragma unroll
    for (int mt = 0; mt < 2; ++mt)
      #pragma unroll
      for (int nt = 0; nt < 4; ++nt)
        acc[mt][nt] = wmma16x16x32(a[mt], b[nt], acc[mt][nt]);
  }

  #pragma unroll
  for (int nt = 0; nt < 4; ++nt) {
    const float bl = bias[nbase + nt * 16 + col];
    #pragma unroll
    for (int mt = 0; mt < 2; ++mt)
      #pragma unroll
      for (int r = 0; r < 8; ++r) {
        const int m = m0 + mt * 16 + r + 8 * kh;
        out[(size_t)m * kE + nbase + nt * 16 + col] = acc[mt][nt][r] + bl;
      }
  }
}

// =====================================================================
extern "C" void kernel_launch(void* const* d_in, const int* in_sizes, int n_in,
                              void* d_out, int out_size, void* d_ws, size_t ws_size,
                              hipStream_t stream) {
  const float* qx   = (const float*)d_in[0];
  const float* kx   = (const float*)d_in[1];
  const float* vx   = (const float*)d_in[2];
  const float* Win  = (const float*)d_in[3];
  const float* bin  = (const float*)d_in[4];
  const float* Wout = (const float*)d_in[5];
  const float* bout = (const float*)d_in[6];
  float* out = (float*)d_out;

  const size_t headElems = (size_t)kB * kH * kS * kHD;   // 8,388,608 per buffer
  _Float16* q0   = (_Float16*)d_ws;
  _Float16* q1   = q0 + headElems;
  _Float16* kbuf = q1 + headElems;
  _Float16* vbuf = kbuf + headElems;
  _Float16* ao   = vbuf + headElems;                     // total 80 MB f16

  // 1) fused QKV projection + bias + layernorm (f32 -> f16 heads)
  proj_ln_kernel<<<dim3(48, 64), 128, 0, stream>>>(qx, kx, vx, Win, bin, q0, kbuf, vbuf);
  // 2) Hopfield updates: q <- LN(softmax(qKᵀ/8) K), twice
  attn_pass_kernel<<<dim3(128, 16), 128, 0, stream>>>(q0, kbuf, kbuf, q1, 0);
  attn_pass_kernel<<<dim3(128, 16), 128, 0, stream>>>(q1, kbuf, kbuf, q0, 0);
  // 3) final attention read-out with V, laid out (S,B,E)
  attn_pass_kernel<<<dim3(128, 16), 128, 0, stream>>>(q0, kbuf, vbuf, ao, 1);
  // 4) output projection + bias -> f32 result
  outproj_kernel<<<dim3(16, 64), 128, 0, stream>>>(ao, Wout, bout, out);
}
